// TinyMoELayer_29661044146275
// MI455X (gfx1250) — matrix operand order repaired and checked by reading, not compiled, not run on hardware
//
#include <hip/hip_runtime.h>
#include <hip/hip_bf16.h>

#define HDIM 1024
#define NEXP 8
#define BTOK 8192
#define MT   32     // token tile per workgroup (2 WMMA A-tiles)

typedef __attribute__((ext_vector_type(16))) __bf16 v16bf;
typedef __attribute__((ext_vector_type(8)))  __bf16 v8bf;
typedef __attribute__((ext_vector_type(8)))  float  v8f;

__device__ __forceinline__ float gelu_exact(float v) {
    return 0.5f * v * (1.0f + erff(v * 0.70710678118654752f));
}

// ---------------------------------------------------------------- zero output
__global__ void moe_zero_kernel(float* __restrict__ p, int n) {
    int i = blockIdx.x * 256 + threadIdx.x;
    if (i < n) p[i] = 0.0f;
}

// ---------------------------------------------------------------- router
__global__ __launch_bounds__(256) void moe_router_kernel(
    const float* __restrict__ inp, const float* __restrict__ Wr,
    const float* __restrict__ br, float* __restrict__ weights_out,
    int* __restrict__ topidx, float* __restrict__ topgate)
{
    int wid  = threadIdx.x >> 5;
    int lane = threadIdx.x & 31;
    int b = blockIdx.x * 8 + wid;

    const float* x = inp + (size_t)b * HDIM;
    float acc[NEXP];
#pragma unroll
    for (int e = 0; e < NEXP; ++e) acc[e] = 0.0f;

    for (int i = 0; i < HDIM / 32; ++i) {
        int h = i * 32 + lane;
        float xv = x[h];
        const float* w = Wr + h * NEXP;
#pragma unroll
        for (int e = 0; e < NEXP; ++e) acc[e] += xv * w[e];
    }
#pragma unroll
    for (int off = 16; off >= 1; off >>= 1) {
#pragma unroll
        for (int e = 0; e < NEXP; ++e)
            acc[e] += __shfl_xor(acc[e], off, 32);
    }

    if (lane == 0) {
        float l[NEXP], w[NEXP];
        float mx = -1e30f;
#pragma unroll
        for (int e = 0; e < NEXP; ++e) { l[e] = acc[e] + br[e]; mx = l[e] > mx ? l[e] : mx; }
        float s = 0.0f;
#pragma unroll
        for (int e = 0; e < NEXP; ++e) { w[e] = __expf(l[e] - mx); s += w[e]; }
        float inv = 1.0f / s;
#pragma unroll
        for (int e = 0; e < NEXP; ++e) { w[e] *= inv; weights_out[b * NEXP + e] = w[e]; }
        int i1 = 0;
#pragma unroll
        for (int e = 1; e < NEXP; ++e) if (w[e] > w[i1]) i1 = e;
        int i2 = (i1 == 0) ? 1 : 0;
#pragma unroll
        for (int e = 0; e < NEXP; ++e) if (e != i1 && w[e] > w[i2]) i2 = e;
        float g1 = w[i1], g2 = w[i2];
        float d = 1.0f / (g1 + g2 + 1e-9f);
        topidx[2 * b]      = i1;
        topidx[2 * b + 1]  = i2;
        topgate[2 * b]     = g1 * d;
        topgate[2 * b + 1] = g2 * d;
    }
}

// ------------------------------------------- f32 [E][K][N] -> bf16 [E][N][K]
__global__ void moe_transpose_bf16_kernel(const float* __restrict__ src,
                                          __bf16* __restrict__ dst)
{
    __shared__ float tile[32][33];
    int e  = blockIdx.z;
    int x0 = blockIdx.x * 32;
    int y0 = blockIdx.y * 32;
    const float* s = src + (size_t)e * HDIM * HDIM;
    __bf16*      d = dst + (size_t)e * HDIM * HDIM;

    for (int i = threadIdx.y; i < 32; i += 8)
        tile[i][threadIdx.x] = s[(size_t)(y0 + i) * HDIM + x0 + threadIdx.x];
    __syncthreads();
    for (int i = threadIdx.y; i < 32; i += 8)
        d[(size_t)(x0 + i) * HDIM + y0 + threadIdx.x] = (__bf16)tile[threadIdx.x][i];
}

// ------------------------------ deterministic per-expert token list (no atomics)
__global__ __launch_bounds__(256) void moe_build_lists_kernel(
    const int* __restrict__ topidx, const float* __restrict__ topgate,
    int* __restrict__ list, float* __restrict__ gatelist, int* __restrict__ counts)
{
    int e = blockIdx.x;
    __shared__ unsigned wtot[8], wpre[8];
    __shared__ unsigned sbase, stot;
    int tid = threadIdx.x, wid = tid >> 5, lane = tid & 31;
    if (tid == 0) sbase = 0;
    __syncthreads();

    for (int c0 = 0; c0 < BTOK; c0 += 256) {
        int t = c0 + tid;
        int i1 = topidx[2 * t], i2 = topidx[2 * t + 1];
        bool flag = (i1 == e) || (i2 == e);
        unsigned mask = (unsigned)__ballot(flag);
        unsigned lp = __popc(mask & ((1u << lane) - 1u));
        if (lane == 0) wtot[wid] = __popc(mask);
        __syncthreads();
        if (tid == 0) {
            unsigned run = sbase;
            for (int w = 0; w < 8; ++w) { wpre[w] = run; run += wtot[w]; }
            stot = run;
        }
        __syncthreads();
        if (flag) {
            unsigned pos = wpre[wid] + lp;
            list[e * BTOK + pos]     = t;
            gatelist[e * BTOK + pos] = (i1 == e) ? topgate[2 * t] : topgate[2 * t + 1];
        }
        __syncthreads();
        if (tid == 0) sbase = stot;
        __syncthreads();
    }
    if (tid == 0) counts[e] = sbase;
}

// ------------------------------------------------- fused two-GEMM expert MLP
// block = 8 waves; (expert, 32-token) tile; K-outer loop, 8 N-tiles/wave inner,
// 16 independent accumulators per wave -> no WMMA dependency stalls.
__global__ __launch_bounds__(256) void moe_gemm_kernel(
    const float*  __restrict__ inp,
    const __bf16* __restrict__ W1t, const float* __restrict__ b1,
    const __bf16* __restrict__ W2t, const float* __restrict__ b2,
    const int*    __restrict__ list, const float* __restrict__ gatelist,
    const int*    __restrict__ counts, float* __restrict__ out)
{
    int e   = blockIdx.y;
    int cnt = counts[e];
    int m0  = blockIdx.x * MT;
    if (m0 >= cnt) return;
    int rows = cnt - m0; if (rows > MT) rows = MT;

    __shared__ __align__(16) __bf16 hs[MT * HDIM];   // 64 KB h tile (bf16)
    __shared__ int    stok[MT];
    __shared__ float  sgate[MT];

    int tid = threadIdx.x;
    if (tid < MT) {
        // padding rows alias token 0: finite garbage, masked at the final write
        stok[tid]  = (tid < rows) ? list[e * BTOK + m0 + tid] : list[e * BTOK + m0];
        sgate[tid] = (tid < rows) ? gatelist[e * BTOK + m0 + tid] : 0.0f;
    }
    __syncthreads();

    int wid = tid >> 5, lane = tid & 31;
    int half = lane >> 4, l15 = lane & 15;
    const __bf16* W1e = W1t + (size_t)e * HDIM * HDIM;
    const __bf16* W2e = W2t + (size_t)e * HDIM * HDIM;

    const float* xr0 = inp + (size_t)stok[l15] * HDIM;        // rows 0..15
    const float* xr1 = inp + (size_t)stok[l15 + 16] * HDIM;   // rows 16..31
    int nbase = wid * 8;   // this wave's 8 N-tiles: columns [nbase*16, nbase*16+128)

    v8f c0[8], c1[8];
    {
        v8f z = {};
#pragma unroll
        for (int j = 0; j < 8; ++j) { c0[j] = z; c1[j] = z; }
    }

    // -------- GEMM1: h = GELU(x @ W1 + b1)
    for (int kb = 0; kb < HDIM; kb += 32) {
        int k1 = kb + half * 16, k2 = kb + half * 16 + 8;
        v16bf a0, a1;
#pragma unroll
        for (int i = 0; i < 8; ++i) {
            a0[i]     = (__bf16)xr0[k1 + i];
            a0[8 + i] = (__bf16)xr0[k2 + i];
            a1[i]     = (__bf16)xr1[k1 + i];
            a1[8 + i] = (__bf16)xr1[k2 + i];
        }
#pragma unroll
        for (int j = 0; j < 8; ++j) {
            const __bf16* wc = W1e + (size_t)((nbase + j) * 16 + l15) * HDIM + kb + half * 16;
            v16bf bfr = *(const v16bf*)wc;
            c0[j] = __builtin_amdgcn_wmma_f32_16x16x32_bf16(
                        false, a0, false, bfr, (short)0, c0[j], false, false);
            c1[j] = __builtin_amdgcn_wmma_f32_16x16x32_bf16(
                        false, a1, false, bfr, (short)0, c1[j], false, false);
        }
    }
    // bias + exact GELU -> LDS h tile
#pragma unroll
    for (int j = 0; j < 8; ++j) {
        int n = (nbase + j) * 16 + l15;
        float bias = b1[e * HDIM + n];
#pragma unroll
        for (int r = 0; r < 8; ++r) {
            int m = half * 8 + r;
            hs[m * HDIM + n]        = (__bf16)gelu_exact(c0[j][r] + bias);
            hs[(m + 16) * HDIM + n] = (__bf16)gelu_exact(c1[j][r] + bias);
        }
    }
    __syncthreads();

    // -------- GEMM2: out += gate * (h @ W2 + b2)
    {
        v8f z = {};
#pragma unroll
        for (int j = 0; j < 8; ++j) { c0[j] = z; c1[j] = z; }
    }
    for (int kb = 0; kb < HDIM; kb += 32) {
        int k1 = kb + half * 16, k2 = kb + half * 16 + 8;
        union { v16bf v; v8bf h[2]; } a0u, a1u;
        a0u.h[0] = *(const v8bf*)(hs + l15 * HDIM + k1);
        a0u.h[1] = *(const v8bf*)(hs + l15 * HDIM + k2);
        a1u.h[0] = *(const v8bf*)(hs + (l15 + 16) * HDIM + k1);
        a1u.h[1] = *(const v8bf*)(hs + (l15 + 16) * HDIM + k2);
#pragma unroll
        for (int j = 0; j < 8; ++j) {
            const __bf16* wc = W2e + (size_t)((nbase + j) * 16 + l15) * HDIM + kb + half * 16;
            v16bf bfr = *(const v16bf*)wc;
            c0[j] = __builtin_amdgcn_wmma_f32_16x16x32_bf16(
                        false, a0u.v, false, bfr, (short)0, c0[j], false, false);
            c1[j] = __builtin_amdgcn_wmma_f32_16x16x32_bf16(
                        false, a1u.v, false, bfr, (short)0, c1[j], false, false);
        }
    }
    // epilogue: bias, gate scale, deterministic atomic combine (2 adds/element)
#pragma unroll
    for (int j = 0; j < 8; ++j) {
        int n = (nbase + j) * 16 + l15;
        float bias = b2[e * HDIM + n];
#pragma unroll
        for (int r = 0; r < 8; ++r) {
            int m = half * 8 + r;
            if (m < rows) {
                float v = (c0[j][r] + bias) * sgate[m];
                unsafeAtomicAdd(&out[(size_t)stok[m] * HDIM + n], v);
            }
            int m2 = m + 16;
            if (m2 < rows) {
                float v = (c1[j][r] + bias) * sgate[m2];
                unsafeAtomicAdd(&out[(size_t)stok[m2] * HDIM + n], v);
            }
        }
    }
}

// ---------------------------------------------------------------- host launch
extern "C" void kernel_launch(void* const* d_in, const int* in_sizes, int n_in,
                              void* d_out, int out_size, void* d_ws, size_t ws_size,
                              hipStream_t stream)
{
    const float* inputs = (const float*)d_in[0];
    const float* Wr     = (const float*)d_in[1];
    const float* br     = (const float*)d_in[2];
    const float* W1     = (const float*)d_in[3];
    const float* b1     = (const float*)d_in[4];
    const float* W2     = (const float*)d_in[5];
    const float* b2     = (const float*)d_in[6];

    float* out         = (float*)d_out;                    // combined [B*H]
    float* weights_out = out + (size_t)BTOK * HDIM;        // weights  [B*E]

    __bf16* W1t      = (__bf16*)d_ws;
    __bf16* W2t      = W1t + (size_t)NEXP * HDIM * HDIM;
    int*    list     = (int*)  (W2t + (size_t)NEXP * HDIM * HDIM);
    float*  gatelist = (float*)(list + NEXP * BTOK);
    int*    topidx   = (int*)  (gatelist + NEXP * BTOK);
    float*  topgate  = (float*)(topidx + 2 * BTOK);
    int*    counts   = (int*)  (topgate + 2 * BTOK);

    {
        int n = BTOK * HDIM;
        moe_zero_kernel<<<(n + 255) / 256, 256, 0, stream>>>(out, n);
    }
    moe_router_kernel<<<BTOK / 8, 256, 0, stream>>>(inputs, Wr, br,
                                                    weights_out, topidx, topgate);
    {
        dim3 g(HDIM / 32, HDIM / 32, NEXP), b(32, 8);
        moe_transpose_bf16_kernel<<<g, b, 0, stream>>>(W1, W1t);
        moe_transpose_bf16_kernel<<<g, b, 0, stream>>>(W2, W2t);
    }
    moe_build_lists_kernel<<<NEXP, 256, 0, stream>>>(topidx, topgate,
                                                     list, gatelist, counts);
    {
        dim3 g(BTOK / MT, NEXP);
        moe_gemm_kernel<<<g, 256, 0, stream>>>(inputs, W1t, b1, W2t, b2,
                                               list, gatelist, counts, out);
    }
}
